// MultiHeadHierarchicalAttention_51479478010208
// MI455X (gfx1250) — matrix operand order repaired and checked by reading, not compiled, not run on hardware
//
#include <hip/hip_runtime.h>
#include <stdint.h>

// ---- WMMA types (gfx1250, wave32) ----
typedef __attribute__((ext_vector_type(16))) __bf16 v16bf;
typedef __attribute__((ext_vector_type(8)))  float  v8f;

union Frag { uint4 u[2]; v16bf v; };

__device__ inline v8f vzero8() {
  v8f z = {0.f, 0.f, 0.f, 0.f, 0.f, 0.f, 0.f, 0.f};
  return z;
}

__device__ inline v8f wmma_bf16(const Frag& a, const Frag& b, v8f c) {
  return __builtin_amdgcn_wmma_f32_16x16x32_bf16(
      /*neg_a=*/false, a.v, /*neg_b=*/false, b.v,
      /*c_mod=*/(short)0, c, /*reuse_a=*/false, /*reuse_b=*/false);
}

// A-fragment (16x32 bf16): lane covers K {kh..kh+7, kh+16..kh+23}, kh=8 for lanes>=16
__device__ inline Frag loadA(const unsigned short* p) {
  Frag f;
  f.u[0] = *(const uint4*)p;
  f.u[1] = *(const uint4*)(p + 16);
  return f;
}
// B-fragment (32x16 bf16): lane covers K {kh..kh+7, kh+16..kh+23}, kh=8 for lanes>=16
__device__ inline Frag loadB(const unsigned short* p) {
  Frag f;
  f.u[0] = *(const uint4*)p;
  f.u[1] = *(const uint4*)(p + 16);
  return f;
}

// f32 -> bf16 round-to-nearest-even
__device__ inline unsigned bfpack2(float a, float b) {
  unsigned ua = __float_as_uint(a), ub = __float_as_uint(b);
  ua = (ua + 0x7FFFu + ((ua >> 16) & 1u)) >> 16;
  ub = (ub + 0x7FFFu + ((ub >> 16) & 1u)) >> 16;
  return (ua & 0xFFFFu) | (ub << 16);
}
__device__ inline unsigned short bf1(float a) {
  unsigned ua = __float_as_uint(a);
  return (unsigned short)((ua + 0x7FFFu + ((ua >> 16) & 1u)) >> 16);
}

#define N_HEADS 8
#define D_IN 512
#define KD 64
#define VD 64
#define SEQ_I 256
#define SEQ_J 256
#define N_T 32
#define N_B 4
#define ATT_SCALE 0.125f  // 1/sqrt(64)

// ---------------------------------------------------------------------------
// Kernel 0: stream-convert f32 -> bf16, 8 elements/thread (b128 in, b128 out)
// ---------------------------------------------------------------------------
__global__ void xconv(const float* __restrict__ X, unsigned short* __restrict__ Xb, long n8) {
  long tid = (long)blockIdx.x * blockDim.x + threadIdx.x;
  if (tid >= n8) return;
  const float4* p = (const float4*)X + tid * 2;
  float4 x0 = p[0], x1 = p[1];
  ((uint4*)Xb)[tid] = make_uint4(bfpack2(x0.x, x0.y), bfpack2(x0.z, x0.w),
                                 bfpack2(x1.x, x1.y), bfpack2(x1.z, x1.w));
}

// ---------------------------------------------------------------------------
// Kernel 1: transpose+convert weights  W[h][d][n] f32 -> Wt[h][n][d] bf16
// ---------------------------------------------------------------------------
__global__ void wprep(const float* __restrict__ W, unsigned short* __restrict__ Wt) {
  int tid = blockIdx.x * blockDim.x + threadIdx.x;
  if (tid >= N_HEADS * KD * D_IN) return;
  int k = tid & (D_IN - 1);         // d index
  int n = (tid >> 9) & (KD - 1);    // kd/vd index
  int hh = tid >> 15;               // head
  Wt[tid] = bf1(W[((long)hh * D_IN + k) * KD + n]);
}

// ---------------------------------------------------------------------------
// Kernel 2: projection GEMM  Out[g][h] = bf16( Xb[g](256x512) @ Wt[h]^T )
//   grid = (mblocks=4, h=8, G), block = 128 (4 waves)
//   wave = 2x2 tile grid: m-tiles {wm*2,wm*2+1}, n-tiles {wn*2,wn*2+1}
//   Software-pipelined: loads for k-step ks+1 precede WMMAs of ks.
// ---------------------------------------------------------------------------
__global__ __launch_bounds__(128) void proj(const unsigned short* __restrict__ Xb, long xg,
                                            const unsigned short* __restrict__ Wt,
                                            const float* __restrict__ bias,
                                            unsigned short* __restrict__ Out,
                                            int transpose) {
  int mblock = blockIdx.x, h = blockIdx.y, g = blockIdx.z;
  int wave = threadIdx.x >> 5, lane = threadIdx.x & 31;
  int ln = lane & 15;
  bool hiL = lane >= 16;
  int wm = wave >> 1, wn = wave & 1;
  int rowbase = mblock * 64;
  int kA = hiL ? 8 : 0, kB = hiL ? 8 : 0;

  const unsigned short* A0 = Xb + (long)g * xg + (long)(rowbase + wm * 32 + ln) * D_IN + kA;
  const unsigned short* A1 = A0 + 16 * D_IN;
  const unsigned short* B0 = Wt + ((long)h * KD + wn * 32 + ln) * D_IN + kB;
  const unsigned short* B1 = B0 + 16 * D_IN;

  v8f c00 = vzero8(), c01 = vzero8(), c10 = vzero8(), c11 = vzero8();

  Frag a0 = loadA(A0), a1 = loadA(A1), b0 = loadB(B0), b1 = loadB(B1);
  for (int ks = 0; ks < D_IN / 32 - 1; ++ks) {
    int kn = (ks + 1) * 32;
    Frag a0n = loadA(A0 + kn), a1n = loadA(A1 + kn);
    Frag b0n = loadB(B0 + kn), b1n = loadB(B1 + kn);
    c00 = wmma_bf16(a0, b0, c00);
    c01 = wmma_bf16(a0, b1, c01);
    c10 = wmma_bf16(a1, b0, c10);
    c11 = wmma_bf16(a1, b1, c11);
    a0 = a0n; a1 = a1n; b0 = b0n; b1 = b1n;
  }
  c00 = wmma_bf16(a0, b0, c00);
  c01 = wmma_bf16(a0, b1, c01);
  c10 = wmma_bf16(a1, b0, c10);
  c11 = wmma_bf16(a1, b1, c11);

  long gh = (long)g * N_HEADS + h;
  int rb = hiL ? 8 : 0;
  v8f accs[4] = {c00, c01, c10, c11};
#pragma unroll
  for (int mi = 0; mi < 2; ++mi) {
#pragma unroll
    for (int ni = 0; ni < 2; ++ni) {
      v8f c = accs[mi * 2 + ni];
      int mrow = rowbase + (wm * 2 + mi) * 16;
      int col = (wn * 2 + ni) * 16 + ln;
      float bv_ = bias[h * KD + col];
      if (!transpose) {
        long base = (gh * 256 + mrow + rb) * KD + col;
#pragma unroll
        for (int r = 0; r < 8; ++r) Out[base + (long)r * KD] = bf1(c[r] + bv_);
      } else {
        long base = (gh * KD + col) * 256 + mrow + rb;
        *(uint4*)(Out + base) = make_uint4(bfpack2(c[0] + bv_, c[1] + bv_),
                                           bfpack2(c[2] + bv_, c[3] + bv_),
                                           bfpack2(c[4] + bv_, c[5] + bv_),
                                           bfpack2(c[6] + bv_, c[7] + bv_));
      }
    }
  }
}

// ---------------------------------------------------------------------------
// Kernel 3: fused two-level attention (online softmax over t).
//   grid = (iblocks=16, h=8, b=4), block = 128 (4 waves)
// ---------------------------------------------------------------------------
__global__ __launch_bounds__(128) void attn(const unsigned short* __restrict__ Q,
                                            const unsigned short* __restrict__ K,
                                            const unsigned short* __restrict__ Vt,
                                            float* __restrict__ Out) {
  int iblock = blockIdx.x, h = blockIdx.y, b = blockIdx.z;
  int tid = threadIdx.x;
  int wave = tid >> 5, lane = tid & 31, ln = lane & 15;
  bool hiL = lane >= 16;

  __shared__ __align__(16) float sS[16][256];            // raw scores
  __shared__ __align__(16) unsigned short sP[16][256];   // exp() as bf16
  __shared__ float sRed[16][8];
  __shared__ float sRowMax[16];
  __shared__ float sM[16], sZ[16], sCA[16], sCB[16];

  // Q fragments for this 16-row i block (kd=64 -> two k-steps), kept in regs
  const unsigned short* Qp = Q + ((long)(b * N_HEADS + h) * SEQ_I + iblock * 16 + ln) * KD;
  int kA = hiL ? 8 : 0, kB = hiL ? 8 : 0;
  Frag aq0 = loadA(Qp + kA);
  Frag aq1 = loadA(Qp + 32 + kA);

  if (tid < 16) { sM[tid] = -1e30f; sZ[tid] = 0.f; }
  v8f acc = vzero8();  // wave's vd tile, online-accumulated over t
  __syncthreads();

  int srow = tid >> 3, seg = tid & 7;
  int rb = hiL ? 8 : 0;

  for (int t = 0; t < N_T; ++t) {
    long bth = (long)((b * N_T + t) * N_HEADS + h);
    const unsigned short* Kp = K + bth * SEQ_J * KD;
    const unsigned short* Vp = Vt + (bth * VD + wave * 16 + ln) * (long)SEQ_J;

    // prefetch next t's K and V blocks (32 KB each; 128 threads * 256B spans it)
    if (t + 1 < N_T) {
      __builtin_prefetch((const char*)(Kp + (long)N_HEADS * SEQ_J * KD) + tid * 256, 0, 3);
      __builtin_prefetch((const char*)(Vt + (bth + N_HEADS) * (long)VD * SEQ_J) + tid * 256, 0, 3);
    }

    // ---- S = Q @ K^T : wave handles j-tiles [wave*4, wave*4+4), pipelined ----
    const unsigned short* krb = Kp + (long)(wave * 64 + ln) * KD;
    Frag kb0 = loadB(krb + kB), kb1 = loadB(krb + 32 + kB);
#pragma unroll
    for (int jt = 0; jt < 4; ++jt) {
      Frag nb0 = kb0, nb1 = kb1;
      if (jt < 3) {
        const unsigned short* krn = krb + (long)(jt + 1) * 16 * KD;
        nb0 = loadB(krn + kB);
        nb1 = loadB(krn + 32 + kB);
      }
      v8f c = vzero8();
      c = wmma_bf16(aq0, kb0, c);
      c = wmma_bf16(aq1, kb1, c);
      int col = wave * 64 + jt * 16 + ln;
#pragma unroll
      for (int r = 0; r < 8; ++r) sS[rb + r][col] = c[r];
      kb0 = nb0; kb1 = nb1;
    }
    __syncthreads();

    // ---- row max over j (raw scores; also the level-2 score s_t) ----
    float m = -1e30f;
#pragma unroll 8
    for (int jj = 0; jj < 32; ++jj) m = fmaxf(m, sS[srow][seg * 32 + jj]);
    sRed[srow][seg] = m;
    __syncthreads();
    if (tid < 16) {
      float rm = sRed[tid][0];
#pragma unroll
      for (int s2 = 1; s2 < 8; ++s2) rm = fmaxf(rm, sRed[tid][s2]);
      sRowMax[tid] = rm;
    }
    __syncthreads();

    // ---- exp + row sum; P (unnormalized) -> bf16 in LDS ----
    float rm = sRowMax[srow];
    float sum = 0.f;
#pragma unroll 8
    for (int jj = 0; jj < 32; ++jj) {
      int j = seg * 32 + jj;
      float e = __expf((sS[srow][j] - rm) * ATT_SCALE);
      sP[srow][j] = bf1(e);
      sum += e;
    }
    sRed[srow][seg] = sum;
    __syncthreads();
    if (tid < 16) {
      float rs = 0.f;
#pragma unroll
      for (int s2 = 0; s2 < 8; ++s2) rs += sRed[tid][s2];
      float st = sRowMax[tid];
      float Mo = sM[tid];
      float Mn = fmaxf(Mo, st);
      float al = __expf((Mo - Mn) * ATT_SCALE);
      float w  = __expf((st - Mn) * ATT_SCALE);
      sZ[tid] = sZ[tid] * al + w;
      sM[tid] = Mn;
      sCA[tid] = al;        // rescale old accumulator
      sCB[tid] = w / rs;    // weight for this t's (unnormalized) retrieved tile
    }
    __syncthreads();

    // ---- O_t = P @ V : wave computes vd-tile `wave`, contraction j=256, pipelined ----
    v8f o = vzero8();
    Frag pa = loadA(&sP[ln][kA]);
    Frag vb = loadB(Vp + kB);
#pragma unroll
    for (int ks = 0; ks < SEQ_J / 32; ++ks) {
      Frag pan = pa, vbn = vb;
      if (ks < SEQ_J / 32 - 1) {
        int kn = (ks + 1) * 32;
        pan = loadA(&sP[ln][kn + kA]);
        vbn = loadB(Vp + kn + kB);
      }
      o = wmma_bf16(pa, vb, o);
      pa = pan; vb = vbn;
    }
#pragma unroll
    for (int r = 0; r < 8; ++r)
      acc[r] = acc[r] * sCA[rb + r] + o[r] * sCB[rb + r];
    __syncthreads();
  }

  // ---- final: divide by Z, store [b][i][vd*h] with col = vd*8 + h ----
  int col = (wave * 16 + ln) * N_HEADS + h;
  long ob = (long)b * SEQ_I + iblock * 16;
#pragma unroll
  for (int r = 0; r < 8; ++r)
    Out[(ob + rb + r) * (VD * N_HEADS) + col] = acc[r] / sZ[rb + r];
}

// ---------------------------------------------------------------------------
extern "C" void kernel_launch(void* const* d_in, const int* in_sizes, int n_in,
                              void* d_out, int out_size, void* d_ws, size_t ws_size,
                              hipStream_t stream) {
  (void)in_sizes; (void)n_in; (void)out_size; (void)ws_size;
  const float* input_seq = (const float*)d_in[0];
  const float* memory_x  = (const float*)d_in[1];
  const float* memory_y  = (const float*)d_in[2];
  const float* Wq = (const float*)d_in[3];
  const float* bq = (const float*)d_in[4];
  const float* Wk = (const float*)d_in[5];
  const float* bk = (const float*)d_in[6];
  const float* Wv = (const float*)d_in[7];
  const float* bv = (const float*)d_in[8];

  // workspace carve-up (bf16 as ushort)
  char* p = (char*)d_ws;
  const size_t wsz  = (size_t)N_HEADS * KD * D_IN * 2;                        // 0.5 MB
  const size_t qsz  = (size_t)N_B * N_HEADS * SEQ_I * KD * 2;                 // 1 MB
  const size_t kvsz = (size_t)N_B * N_T * N_HEADS * SEQ_J * KD * 2;           // 32 MB
  const size_t xqel = (size_t)N_B * SEQ_I * D_IN;                             // input_seq elems
  const size_t xmel = (size_t)N_B * N_T * SEQ_J * D_IN;                       // memory elems
  unsigned short* WqT = (unsigned short*)p; p += wsz;
  unsigned short* WkT = (unsigned short*)p; p += wsz;
  unsigned short* WvT = (unsigned short*)p; p += wsz;
  unsigned short* Qb  = (unsigned short*)p; p += qsz;
  unsigned short* Kb  = (unsigned short*)p; p += kvsz;
  unsigned short* Vtb = (unsigned short*)p; p += kvsz;
  unsigned short* Xq  = (unsigned short*)p; p += xqel * 2;
  unsigned short* Xm  = (unsigned short*)p;  // 33.5 MB, reused for memory_x then memory_y

  // 1) weight transpose+convert, activation f32->bf16 streams
  int welems = N_HEADS * KD * D_IN;
  wprep<<<(welems + 255) / 256, 256, 0, stream>>>(Wq, WqT);
  wprep<<<(welems + 255) / 256, 256, 0, stream>>>(Wk, WkT);
  wprep<<<(welems + 255) / 256, 256, 0, stream>>>(Wv, WvT);
  xconv<<<(int)((xqel / 8 + 255) / 256), 256, 0, stream>>>(input_seq, Xq, (long)(xqel / 8));

  long xg = (long)SEQ_I * D_IN;  // elements per g for both input_seq and memory

  // 2a) Q projection
  proj<<<dim3(4, N_HEADS, N_B), 128, 0, stream>>>(Xq, xg, WqT, bq, Qb, 0);

  // 2b) K projection (stage memory_x -> bf16, then GEMM)
  xconv<<<(int)((xmel / 8 + 255) / 256), 256, 0, stream>>>(memory_x, Xm, (long)(xmel / 8));
  proj<<<dim3(4, N_HEADS, N_B * N_T), 128, 0, stream>>>(Xm, xg, WkT, bk, Kb, 0);

  // 2c) V projection, transposed output (reuse Xm after K proj has consumed it)
  xconv<<<(int)((xmel / 8 + 255) / 256), 256, 0, stream>>>(memory_y, Xm, (long)(xmel / 8));
  proj<<<dim3(4, N_HEADS, N_B * N_T), 128, 0, stream>>>(Xm, xg, WvT, bv, Vtb, 1);

  // 3) fused hierarchical attention
  attn<<<dim3(SEQ_I / 16, N_HEADS, N_B), 128, 0, stream>>>(Qb, Kb, Vtb, (float*)d_out);
}
